// HungarianMatcher_66726611910810
// MI455X (gfx1250) — matrix operand order repaired and checked by reading, not compiled, not run on hardware
//
#include <hip/hip_runtime.h>
#include <cstdint>

// Problem constants (from reference)
constexpr int kBS   = 2;
constexpr int kQ    = 200;
constexpr int kT    = 80;
constexpr int kNPTS = 200000;
constexpr int kNSUB = 100000;
constexpr int kNCLS = 3;
constexpr int kKCH  = 800;           // K-slice per GEMM workgroup (800/32 = 25 chunks)

// Workspace layout (float words, per batch)
constexpr int O_M1   = 0;            // 200*80
constexpr int O_M2   = 16000;        // 200*80
constexpr int O_NEG  = 32000;        // 200  : sum softplus(x) over n
constexpr int O_SSUM = 32200;        // 200  : sum sigmoid(x) over n
constexpr int O_CNT  = 32400;        // 200  : segment counts
constexpr int O_SUMS = 32600;        // 200*3: segment coord sums
constexpr int O_PMIN = 33200;        // 200*3 (uint keys)
constexpr int O_PMAX = 33800;        // 200*3 (uint keys)
constexpr int O_TCNT = 34400;        // 80
constexpr int O_TSUM = 34480;        // 80*3
constexpr int O_GMIN = 34720;        // 80*3 (uint keys)
constexpr int O_GMAX = 34960;        // 80*3 (uint keys)
constexpr int WSB    = 35200;        // words per batch

typedef __attribute__((ext_vector_type(16))) _Float16 v16h;
typedef __attribute__((ext_vector_type(8)))  float    v8f;

#if defined(__HIP_DEVICE_COMPILE__) && defined(__has_builtin)
#if __has_builtin(__builtin_amdgcn_global_load_async_to_lds_b32) && \
    __has_builtin(__builtin_amdgcn_s_wait_asynccnt)
#define USE_ASYNC_LDS 1
#endif
#endif

typedef __attribute__((address_space(1))) int g_int;
typedef __attribute__((address_space(3))) int l_int;

// Order-preserving float<->uint key (non-NaN): enables atomicMin/Max on u32.
__device__ __forceinline__ unsigned encf(float f) {
  unsigned b = __float_as_uint(f);
  return (b & 0x80000000u) ? ~b : (b | 0x80000000u);
}
__device__ __forceinline__ float decf(unsigned k) {
  unsigned b = (k & 0x80000000u) ? (k & 0x7FFFFFFFu) : ~k;
  return __uint_as_float(b);
}

__global__ void k_init(float* ws) {
  int i = blockIdx.x * blockDim.x + threadIdx.x;
  if (i >= kBS * WSB) return;
  int w = i % WSB;
  unsigned* uw = (unsigned*)ws;
  bool isMin = (w >= O_PMIN && w < O_PMIN + 600) || (w >= O_GMIN && w < O_GMIN + 240);
  bool isMax = (w >= O_PMAX && w < O_PMAX + 600) || (w >= O_GMAX && w < O_GMAX + 240);
  if (isMin)      uw[i] = 0xFF800000u;   // enc(+inf): identity for min
  else if (isMax) uw[i] = 0x007FFFFFu;   // enc(-inf): identity for max
  else            ws[i] = 0.0f;
}

// Pass 1: one wave32 per subsampled point.
// Computes negsum/ssum row reductions, argmax-q segment stats, target stats.
__global__ __launch_bounds__(256) void k_points(const float* __restrict__ pm,
                                                const float* __restrict__ tm,
                                                const float* __restrict__ coords,
                                                const int*   __restrict__ pidx,
                                                float* ws) {
  __shared__ float    sneg[kQ], ssig[kQ], scnt[kQ], ssums[kQ * 3];
  __shared__ unsigned smin[kQ * 3], smax[kQ * 3];
  __shared__ float    stcnt[kT], stsum[kT * 3];
  __shared__ unsigned sgmin[kT * 3], sgmax[kT * 3];

  const int tid = threadIdx.x;
  for (int i = tid; i < kQ; i += 256) { sneg[i] = 0.f; ssig[i] = 0.f; scnt[i] = 0.f; }
  for (int i = tid; i < kQ * 3; i += 256) { ssums[i] = 0.f; smin[i] = 0xFF800000u; smax[i] = 0x007FFFFFu; }
  for (int i = tid; i < kT; i += 256) stcnt[i] = 0.f;
  for (int i = tid; i < kT * 3; i += 256) { stsum[i] = 0.f; sgmin[i] = 0xFF800000u; sgmax[i] = 0x007FFFFFu; }
  __syncthreads();

  const int wave = tid >> 5, lane = tid & 31;
  const int bpb  = kNSUB / 8;                 // blocks per batch (100000/8 = 12500)
  const int b    = blockIdx.x / bpb;
  const int n    = (blockIdx.x % bpb) * 8 + wave;
  const int idx  = pidx[b * kNSUB + n];

  const float* xrow = pm + ((size_t)b * kNPTS + idx) * kQ;
  const float c0 = coords[n * 3 + 0], c1 = coords[n * 3 + 1], c2 = coords[n * 3 + 2];

  float bv = -__builtin_huge_valf(); int bq = 0;
  #pragma unroll
  for (int j = 0; j < 7; j++) {
    int q = lane + 32 * j;
    if (q < kQ) {
      float x  = xrow[q];
      float sp = fmaxf(x, 0.f) + log1pf(__expf(-fabsf(x)));  // softplus(x)
      float sg = 1.f / (1.f + __expf(-x));
      atomicAdd(&sneg[q], sp);
      atomicAdd(&ssig[q], sg);
      if (x > bv) { bv = x; bq = q; }                        // first max within lane
    }
  }
  // wave reduce: max value, smallest q on ties (matches jnp.argmax)
  for (int off = 16; off; off >>= 1) {
    float ov = __shfl_xor(bv, off, 32);
    int   oq = __shfl_xor(bq, off, 32);
    if (ov > bv || (ov == bv && oq < bq)) { bv = ov; bq = oq; }
  }
  if (lane == 0) {
    atomicAdd(&scnt[bq], 1.f);
    atomicAdd(&ssums[bq * 3 + 0], c0);
    atomicAdd(&ssums[bq * 3 + 1], c1);
    atomicAdd(&ssums[bq * 3 + 2], c2);
    unsigned e0 = encf(c0), e1 = encf(c1), e2 = encf(c2);
    atomicMin(&smin[bq * 3 + 0], e0); atomicMax(&smax[bq * 3 + 0], e0);
    atomicMin(&smin[bq * 3 + 1], e1); atomicMax(&smax[bq * 3 + 1], e1);
    atomicMin(&smin[bq * 3 + 2], e2); atomicMax(&smax[bq * 3 + 2], e2);
  }

  // target stats: tgt[t, idx] over lanes
  const float* trow = tm + (size_t)b * kT * kNPTS + idx;
  #pragma unroll
  for (int j = 0; j < 3; j++) {
    int t = lane + 32 * j;
    if (t < kT) {
      float tv = trow[(size_t)t * kNPTS];
      if (tv > 0.f) {
        atomicAdd(&stcnt[t], tv);
        atomicAdd(&stsum[t * 3 + 0], tv * c0);
        atomicAdd(&stsum[t * 3 + 1], tv * c1);
        atomicAdd(&stsum[t * 3 + 2], tv * c2);
        unsigned e0 = encf(c0), e1 = encf(c1), e2 = encf(c2);
        atomicMin(&sgmin[t * 3 + 0], e0); atomicMax(&sgmax[t * 3 + 0], e0);
        atomicMin(&sgmin[t * 3 + 1], e1); atomicMax(&sgmax[t * 3 + 1], e1);
        atomicMin(&sgmin[t * 3 + 2], e2); atomicMax(&sgmax[t * 3 + 2], e2);
      }
    }
  }
  __syncthreads();

  // flush block accumulators
  float*    W  = ws + (size_t)b * WSB;
  unsigned* WU = (unsigned*)W;
  for (int i = tid; i < kQ; i += 256) {
    atomicAdd(&W[O_NEG + i],  sneg[i]);
    atomicAdd(&W[O_SSUM + i], ssig[i]);
    atomicAdd(&W[O_CNT + i],  scnt[i]);
  }
  for (int i = tid; i < kQ * 3; i += 256) {
    atomicAdd(&W[O_SUMS + i], ssums[i]);
    atomicMin(&WU[O_PMIN + i], smin[i]);
    atomicMax(&WU[O_PMAX + i], smax[i]);
  }
  for (int i = tid; i < kT; i += 256) atomicAdd(&W[O_TCNT + i], stcnt[i]);
  for (int i = tid; i < kT * 3; i += 256) {
    atomicAdd(&W[O_TSUM + i], stsum[i]);
    atomicMin(&WU[O_GMIN + i], sgmin[i]);
    atomicMax(&WU[O_GMAX + i], sgmax[i]);
  }
}

// Pass 2: WMMA GEMMs M1 = x @ tgt^T, M2 = sigmoid(x) @ tgt^T over gathered K.
// One 5-wave workgroup per (q-tile, K-slice); the 5 waves cover the 5 t-tiles
// and share one LDS staging of the A chunk (16x32) and tgt chunk (32x80).
__global__ __launch_bounds__(160) void k_gemm(const float* __restrict__ pm,
                                              const float* __restrict__ tm,
                                              const int*   __restrict__ pidx,
                                              float* ws) {
  __shared__ _Float16 Xs[16][32];     // A chunk: x[q0+j][k]
  __shared__ _Float16 Ss[16][32];     // A chunk: sigmoid(x)
  __shared__ _Float16 Ts[32][kT];     // tgt chunk: [k][t]
  __shared__ int idxs[32];

  const int tid   = threadIdx.x;
  const int wave  = tid >> 5;
  const int lane  = tid & 31;
  const int b     = blockIdx.z;
  const int qtile = blockIdx.y;
  const int q0    = qtile * 16;
  const int t0    = wave * 16;        // each wave owns one t-tile
  const int kBase = blockIdx.x * kKCH;

  v8f accX = {}; v8f accS = {};

  const size_t pmB = (size_t)b * kNPTS;
  const int    m   = lane & 15;
  const int    kbA = (lane < 16) ? 0 : 8;    // ISA 16-bit A layout
  const int    kbB = (lane < 16) ? 0 : 8;    // ISA 16-bit B layout
  const int    tcol = t0 + (lane & 15);

  for (int kc = 0; kc < kKCH; kc += 32) {
    // --- stage the 32 gather indices (async copy to LDS when available) ---
    if (tid < 32) {
#ifdef USE_ASYNC_LDS
      __builtin_amdgcn_global_load_async_to_lds_b32(
          (g_int*)(pidx + b * kNSUB + kBase + kc + tid),
          (l_int*)&idxs[tid], 0, 0);
#else
      idxs[tid] = pidx[b * kNSUB + kBase + kc + tid];
#endif
      if (kc + 32 < kKCH)
        __builtin_prefetch(&pidx[b * kNSUB + kBase + kc + tid + 32], 0, 1);
    }
#ifdef USE_ASYNC_LDS
    __builtin_amdgcn_s_wait_asynccnt(0);
#endif
    __syncthreads();

    // --- cooperative A staging: 16 q x 32 k, float4 loads (rows 16B-aligned) ---
    for (int e = tid; e < 128; e += 160) {
      int k  = e & 31;
      int j0 = (e >> 5) * 4;
      const float* xr = pm + (pmB + idxs[k]) * kQ + q0 + j0;
      float x0, x1, x2, x3;
      if (q0 + j0 + 3 < kQ) {
        float4 v = *(const float4*)xr;
        x0 = v.x; x1 = v.y; x2 = v.z; x3 = v.w;
      } else {
        x0 = (q0 + j0 + 0 < kQ) ? xr[0] : 0.f;
        x1 = (q0 + j0 + 1 < kQ) ? xr[1] : 0.f;
        x2 = (q0 + j0 + 2 < kQ) ? xr[2] : 0.f;
        x3 = (q0 + j0 + 3 < kQ) ? xr[3] : 0.f;
      }
      Xs[j0 + 0][k] = (_Float16)x0;
      Xs[j0 + 1][k] = (_Float16)x1;
      Xs[j0 + 2][k] = (_Float16)x2;
      Xs[j0 + 3][k] = (_Float16)x3;
      Ss[j0 + 0][k] = (q0 + j0 + 0 < kQ) ? (_Float16)(1.f / (1.f + __expf(-x0))) : (_Float16)0.f;
      Ss[j0 + 1][k] = (q0 + j0 + 1 < kQ) ? (_Float16)(1.f / (1.f + __expf(-x1))) : (_Float16)0.f;
      Ss[j0 + 2][k] = (q0 + j0 + 2 < kQ) ? (_Float16)(1.f / (1.f + __expf(-x2))) : (_Float16)0.f;
      Ss[j0 + 3][k] = (q0 + j0 + 3 < kQ) ? (_Float16)(1.f / (1.f + __expf(-x3))) : (_Float16)0.f;
    }
    // --- cooperative tgt staging: 32 k x 80 t scattered gathers ---
    for (int e = tid; e < 32 * kT; e += 160) {
      int t = e % kT;
      int k = e / kT;
      Ts[k][t] = (_Float16)tm[((size_t)b * kT + t) * (size_t)kNPTS + idxs[k]];
    }
    __syncthreads();

    // --- build fragments and issue the two WMMAs (shared B) ---
    v16h aX, aS, bT;
    #pragma unroll
    for (int v = 0; v < 8; v++) {
      int k = (v < 4) ? (kbA + 2 * v) : (kbA + 16 + 2 * (v - 4));
      aX[2 * v]     = Xs[m][k];     aX[2 * v + 1] = Xs[m][k + 1];
      aS[2 * v]     = Ss[m][k];     aS[2 * v + 1] = Ss[m][k + 1];
      int kk = (v < 4) ? (kbB + 2 * v) : (kbB + 16 + 2 * (v - 4));
      bT[2 * v]     = Ts[kk][tcol];
      bT[2 * v + 1] = Ts[kk + 1][tcol];
    }
    accX = __builtin_amdgcn_wmma_f32_16x16x32_f16(false, aX, false, bT, (short)0, accX, false, false);
    accS = __builtin_amdgcn_wmma_f32_16x16x32_f16(false, aS, false, bT, (short)0, accS, false, false);
    __syncthreads();
  }

  float* W = ws + (size_t)b * WSB;
  const int nCol = lane & 15;
  const int mAdd = (lane < 16) ? 0 : 8;        // C/D layout: lanes 16-31 hold M+8
  #pragma unroll
  for (int r = 0; r < 8; r++) {
    int q = q0 + r + mAdd;
    int t = t0 + nCol;
    if (q < kQ) {
      atomicAdd(&W[O_M1 + q * kT + t], accX[r]);
      atomicAdd(&W[O_M2 + q * kT + t], accS[r]);
    }
  }
}

// Pass 3: combine all five cost terms.
__global__ void k_final(const float* __restrict__ logits,
                        const int*   __restrict__ tcls,
                        const float* __restrict__ ws,
                        float* __restrict__ out) {
  int i = blockIdx.x * blockDim.x + threadIdx.x;
  if (i >= kBS * kQ * kT) return;
  const int t = i % kT;
  const int q = (i / kT) % kQ;
  const int b = i / (kQ * kT);
  const float*    W  = ws + (size_t)b * WSB;
  const unsigned* WU = (const unsigned*)W;

  // cost_class = -softmax(logits)[cls]
  const float* lg = logits + ((size_t)b * kQ + q) * kNCLS;
  float l0 = lg[0], l1 = lg[1], l2 = lg[2];
  float mx = fmaxf(l0, fmaxf(l1, l2));
  float e0 = __expf(l0 - mx), e1 = __expf(l1 - mx), e2 = __expf(l2 - mx);
  int   cls = tcls[b * kT + t];
  float cost_class = -((cls == 0 ? e0 : (cls == 1 ? e1 : e2)) / (e0 + e1 + e2));

  // cost_mask = (negsum - x@tgt) / NSUB
  float M1 = W[O_M1 + q * kT + t];
  float M2 = W[O_M2 + q * kT + t];
  float cost_mask = (W[O_NEG + q] - M1) * (1.0f / (float)kNSUB);

  // cost_dice
  float denom = W[O_SSUM + q] + W[O_TCNT + t];
  float cost_dice = 1.f - (2.f * M2 + 1.f) / (denom + 1.f);

  // box + center
  float cnt = W[O_CNT + q];
  bool  ne  = cnt > 0.f;
  float inv = 1.f / fmaxf(cnt, 1.f);
  float tc   = W[O_TCNT + t];
  bool  tne  = tc > 0.f;
  float tinv = 1.f / fmaxf(tc, 1.f);
  float dmin = 0.f, dmax = 0.f, dctr = 0.f;
  #pragma unroll
  for (int d = 0; d < 3; d++) {
    float cp   = ne  ? W[O_SUMS + q * 3 + d] * inv       : 0.f;
    float pmin = ne  ? decf(WU[O_PMIN + q * 3 + d])      : 0.f;
    float pmax = ne  ? decf(WU[O_PMAX + q * 3 + d])      : 0.f;
    float cg   = tne ? W[O_TSUM + t * 3 + d] * tinv      : 0.f;
    float gmin = tne ? decf(WU[O_GMIN + t * 3 + d])      : 0.f;
    float gmax = tne ? decf(WU[O_GMAX + t * 3 + d])      : 0.f;
    dmin += (pmin - gmin) * (pmin - gmin);
    dmax += (pmax - gmax) * (pmax - gmax);
    dctr += (cp - cg) * (cp - cg);
  }
  float cost_box    = sqrtf(dmin) + sqrtf(dmax);
  float cost_center = sqrtf(dctr);

  out[i] = 5.f * cost_mask + 2.f * cost_class + 2.f * cost_dice
         + 1.f * cost_box + 1.f * cost_center;
}

extern "C" void kernel_launch(void* const* d_in, const int* in_sizes, int n_in,
                              void* d_out, int out_size, void* d_ws, size_t ws_size,
                              hipStream_t stream) {
  (void)in_sizes; (void)n_in; (void)out_size; (void)ws_size;
  const float* logits = (const float*)d_in[0];   // (BS,Q,NCLS)
  const float* pm     = (const float*)d_in[1];   // (BS,NPTS,Q)
  const float* tm     = (const float*)d_in[2];   // (BS,T,NPTS)
  const float* coords = (const float*)d_in[3];   // (NPTS,3)
  const int*   tcls   = (const int*)d_in[4];     // (BS,T)
  const int*   pidx   = (const int*)d_in[5];     // (BS,NSUB)
  float* out = (float*)d_out;                    // (BS,Q,T)
  float* ws  = (float*)d_ws;                     // >= 2*WSB*4 bytes (~282 KB)

  const int initN = kBS * WSB;
  k_init<<<(initN + 255) / 256, 256, 0, stream>>>(ws);

  k_points<<<kBS * (kNSUB / 8), 256, 0, stream>>>(pm, tm, coords, pidx, ws);

  dim3 gg(kNSUB / kKCH, 13, kBS);                // (125, 13, 2), 5 waves each
  k_gemm<<<gg, 160, 0, stream>>>(pm, tm, pidx, ws);

  const int fn = kBS * kQ * kT;
  k_final<<<(fn + 255) / 256, 256, 0, stream>>>(logits, tcls, ws, out);
}
